// Seq2SeqDiffusion_82351702934308
// MI455X (gfx1250) — compile-verified, matrix operand order checked by reading, not run on hardware
//
#include <hip/hip_runtime.h>
#include <hip/hip_bf16.h>

// ---------------------------------------------------------------------------
// Persistent-kernel seq2seq LSTM for MI455X (gfx1250, wave32, WMMA).
// Round 2: interleaved gate-column ownership so the LSTM cell update is
// workgroup-local:
//   * WG g owns gate columns { gate*512 + 4g + (0..3) } for gate = i,f,g,o,
//     i.e. all four gates of hidden units [4g, 4g+4).
//   * gemm -> cell needs only a block fence (grid barriers: 2592 -> 1440).
//   * Cell state c lives in LDS (8 KB, WG-exclusive); gate tile is a
//     WG-private 16 KB global scratch (no cross-WG coherence).
//   * Decoder projection = per-WG partial dots + fixed-order reduction in
//     WG0 (deterministic, no FP atomics).
//   * Weights bf16, pre-swizzled B-fragments resident in LDS (50 KB),
//     restaged once at the enc->dec transition. v_wmma_f32_16x16x32_bf16,
//     f32 accumulation.
// ---------------------------------------------------------------------------

typedef __attribute__((ext_vector_type(16))) __bf16 v16bf;
typedef __attribute__((ext_vector_type(8)))  float  v8f;

#define NBLK   128
#define TPB    256
#define BATCH  256
#define HID    512
#define G4H    2048
#define TSTEPS 288

union ABu { v16bf v; unsigned u[8]; };
union CDu { v8f   v; float    f[8]; };

__device__ __forceinline__ unsigned lane_id() { return threadIdx.x & 31u; }

// WG `wg`, tile column j (0..15) -> weight-matrix row (gate column).
__device__ __forceinline__ int colmap(int j, int wg) {
  return (j >> 2) * 512 + wg * 4 + (j & 3);
}

// A fragment: 16x32 bf16, rows b0..b0+15 of row-major src[*][ldK], cols kc*32..+31.
// ISA 16-bit A layout: lanes 0-15 hold M=lane, lanes 16-31 same M, upper K-half;
// VGPR v: kbase = (v>>2)*16 + (lane>>4)*8 + (v&3)*2, packed pair (k,k+1).
__device__ __forceinline__ v16bf load_a(const __bf16* src, int b0, int ldK, int kc) {
  ABu r;
  unsigned l = lane_id();
  int m = l & 15, half = l >> 4;
  const __bf16* row = src + (size_t)(b0 + m) * ldK + kc * 32;
#pragma unroll
  for (int v = 0; v < 8; ++v) {
    int kbase = (v >> 2) * 16 + half * 8 + (v & 3) * 2;
    r.u[v] = *(const unsigned*)(row + kbase);   // packed bf16 pair, 4B aligned
  }
  return r.v;
}

// B fragment from LDS, pre-swizzled: ldsW[chunk][lane][v] (uints).
__device__ __forceinline__ v16bf load_b(const unsigned* ldsW, int chunk) {
  ABu r;
  const unsigned* p = ldsW + ((chunk << 5) + lane_id()) * 8;
#pragma unroll
  for (int v = 0; v < 8; ++v) r.u[v] = p[v];
  return r.v;
}

struct WMat { const __bf16* w; int K; int nch; };

// Pre-swizzle this WG's 16 (interleaved) weight columns into LDS B-fragments.
__device__ void stage_weights(unsigned* ldsW, const WMat* mats, int nmat, int wg) {
  int base = 0;
  for (int mi = 0; mi < nmat; ++mi) {
    const __bf16* W = mats[mi].w;
    int K = mats[mi].K;
    int tot = mats[mi].nch * 256;
    for (int idx = threadIdx.x; idx < tot; idx += TPB) {
      int v = idx & 7, l = (idx >> 3) & 31, c = idx >> 8;
      int n = colmap(l & 15, wg), half = l >> 4;
      int kbase = (v >> 2) * 16 + half * 8 + (v & 3) * 2;
      int k = c * 32 + kbase;
      // W row-major [2048][K] bf16: pair (k,k+1) contiguous -> one 32-bit load
      ldsW[((base + c) << 8) + l * 8 + v] = *(const unsigned*)(W + (size_t)n * K + k);
    }
    base += mats[mi].nch;
  }
  __syncthreads();
}

// Fused GEMM into WG-private compact gate tile Gloc[256][16] (f32).
__device__ __forceinline__ void gemm_phase(
    const __bf16* A0, int nch0, int ldK0, int cb0,
    const __bf16* A1, int nch1, int ldK1, int cb1,
    const unsigned* ldsW, float* Gloc) {
  int wave = threadIdx.x >> 5;
#pragma unroll
  for (int rt = 0; rt < 2; ++rt) {
    int b0 = (wave * 2 + rt) * 16;
    CDu acc;
#pragma unroll
    for (int r = 0; r < 8; ++r) acc.f[r] = 0.f;
    for (int c = 0; c < nch0; ++c) {
      v16bf a = load_a(A0, b0, ldK0, c);
      v16bf b = load_b(ldsW, cb0 + c);
      acc.v = __builtin_amdgcn_wmma_f32_16x16x32_bf16(false, a, false, b,
                                                      (short)0, acc.v, false, false);
    }
    for (int c = 0; c < nch1; ++c) {
      v16bf a = load_a(A1, b0, ldK1, c);
      v16bf b = load_b(ldsW, cb1 + c);
      acc.v = __builtin_amdgcn_wmma_f32_16x16x32_bf16(false, a, false, b,
                                                      (short)0, acc.v, false, false);
    }
    // C/D layout: VGPR r -> M = r + 8*(lane>>4), N = lane&15 (tile-local col j)
    unsigned l = lane_id();
    int j = l & 15, mh = (l >> 4) * 8;
#pragma unroll
    for (int r = 0; r < 8; ++r)
      Gloc[(b0 + mh + r) * 16 + j] = acc.f[r];   // 64B-contiguous row stores
  }
}

__device__ __forceinline__ float sigf(float x) { return 1.f / (1.f + __expf(-x)); }

// WG-local LSTM cell update for hidden units [col0, col0+4) x all 256 rows.
// cL: LDS c-state slice [4][256] (hh-major, conflict-free).
// If projW != nullptr: also emit this WG's partial projection dot per row.
__device__ void cell_local(const float* Gloc, const float* bias, float* cL,
                           __bf16* hbf, int col0,
                           const float* projW, float* ypartRow) {
  int b = threadIdx.x;                     // 256 threads == 256 batch rows
  const float* gr = Gloc + b * 16;
  float ysum = 0.f;
#pragma unroll
  for (int hh = 0; hh < 4; ++hh) {
    int h = col0 + hh;
    float gi = gr[hh]      + bias[h];
    float gf = gr[4 + hh]  + bias[512 + h];
    float gg = gr[8 + hh]  + bias[1024 + h];
    float go = gr[12 + hh] + bias[1536 + h];
    int ci = hh * 256 + b;
    float c2 = sigf(gf) * cL[ci] + sigf(gi) * tanhf(gg);
    cL[ci] = c2;
    float hv = sigf(go) * tanhf(c2);
    hbf[(size_t)b * HID + h] = (__bf16)hv;
    if (projW) ysum += hv * projW[h];
  }
  if (ypartRow) ypartRow[b] = ysum;
}

// Epoch-counter grid barrier (counter zeroed each launch by prep kernel).
__device__ __forceinline__ void gsync(unsigned* bar, unsigned goal) {
  __syncthreads();
  if (threadIdx.x == 0) {
    __threadfence();
    __hip_atomic_fetch_add(bar, 1u, __ATOMIC_ACQ_REL, __HIP_MEMORY_SCOPE_AGENT);
    while (__hip_atomic_load(bar, __ATOMIC_ACQUIRE, __HIP_MEMORY_SCOPE_AGENT) < goal)
      __builtin_amdgcn_s_sleep(2);
    __threadfence();
  }
  __syncthreads();
}

__device__ __forceinline__ void block_sync() {
  __threadfence_block();
  __syncthreads();
}

struct Params {
  const __bf16 *encWih0p, *encWhh0, *encWih1, *encWhh1;
  const __bf16 *decWih0p, *decWhh0, *decWih1, *decWhh1;
  const __bf16 *encX;
  __bf16 *decX, *h0bf, *h1bf;
  float *G, *ypart;
  const float *encB0, *encB1, *decB0, *decB1, *projW, *projB;
  float *out;
  unsigned *bar;
};

__global__ __launch_bounds__(TPB, 1) void k_lstm(Params p) {
  __shared__ unsigned ldsW[49 * 256];           // 49 chunks * 1KB = 50176 B
  __shared__ float    ldsC[2 * 1024];           // c-state, 2 layers x 4h x 256b
  int wg = blockIdx.x;
  int col0 = wg * 4;
  float* Gloc = p.G + (size_t)wg * BATCH * 16;  // WG-private gate tile
  float* ypartRow = p.ypart + (size_t)wg * BATCH;
  unsigned ep = 0;

  for (int i = threadIdx.x; i < 2 * 1024; i += TPB) ldsC[i] = 0.f;

  { WMat m[4] = {{p.encWih0p, 32, 1}, {p.encWhh0, 512, 16},
                 {p.encWih1, 512, 16}, {p.encWhh1, 512, 16}};
    stage_weights(ldsW, m, 4, wg); }            // chunk bases: 0,1,17,33

  for (int t = 0; t < TSTEPS; ++t) {            // -------- encoder --------
    gemm_phase(p.encX + (size_t)t * BATCH * 32, 1, 32, 0,
               p.h0bf, 16, 512, 1, ldsW, Gloc);
    block_sync();
    cell_local(Gloc, p.encB0, ldsC, p.h0bf, col0, nullptr, nullptr);
    gsync(p.bar, ++ep * NBLK);                  // h0 visible device-wide
    gemm_phase(p.h0bf, 16, 512, 17, p.h1bf, 16, 512, 33, ldsW, Gloc);
    block_sync();
    cell_local(Gloc, p.encB1, ldsC + 1024, p.h1bf, col0, nullptr, nullptr);
    gsync(p.bar, ++ep * NBLK);                  // h1 visible device-wide
  }

  __syncthreads();
  { WMat m[4] = {{p.decWih0p, 32, 1}, {p.decWhh0, 512, 16},
                 {p.decWih1, 512, 16}, {p.decWhh1, 512, 16}};
    stage_weights(ldsW, m, 4, wg); }

  for (int t = 0; t < TSTEPS; ++t) {            // -------- decoder --------
    gemm_phase(p.decX + (size_t)t * BATCH * 32, 1, 32, 0,
               p.h0bf, 16, 512, 1, ldsW, Gloc);
    block_sync();
    cell_local(Gloc, p.decB0, ldsC, p.h0bf, col0, nullptr, nullptr);
    gsync(p.bar, ++ep * NBLK);
    gemm_phase(p.h0bf, 16, 512, 17, p.h1bf, 16, 512, 33, ldsW, Gloc);
    block_sync();
    cell_local(Gloc, p.decB1, ldsC + 1024, p.h1bf, col0, p.projW, ypartRow);
    gsync(p.bar, ++ep * NBLK);                  // h1 + ypart visible
    if (wg == 0) {                              // fixed-order projection reduce
      int b = threadIdx.x;
      float s = p.projB[0];
      for (int g = 0; g < NBLK; ++g) s += p.ypart[(size_t)g * BATCH + b];
      p.out[(size_t)b * TSTEPS + t] = s;
      if (t + 1 < TSTEPS)
        p.decX[(size_t)(t + 1) * BATCH * 32 + b * 32] = (__bf16)s;
    }
    gsync(p.bar, ++ep * NBLK);                  // y feedback visible
  }
}

// ------------------------------- prep kernels ------------------------------

__global__ void k_cvt_pad(const float* src, __bf16* dst, int rows, int Ks, int Kd) {
  int tot = rows * Kd;
  for (int i = blockIdx.x * TPB + threadIdx.x; i < tot; i += gridDim.x * TPB) {
    int r = i / Kd, k = i - r * Kd;
    dst[i] = (__bf16)((k < Ks) ? src[(size_t)r * Ks + k] : 0.f);
  }
}

__global__ void k_add(const float* a, const float* b, float* o, int n) {
  int i = blockIdx.x * TPB + threadIdx.x;
  if (i < n) o[i] = a[i] + b[i];
}

__global__ void k_build_encx(const float* xh, const float* emb, const int* tidx, __bf16* X) {
  int i = blockIdx.x * TPB + threadIdx.x;       // over T*B
  if (i >= TSTEPS * BATCH) return;
  int t = i / BATCH, b = i - t * BATCH;
  __bf16* d = X + (size_t)i * 32;
  const float* e = emb + (size_t)tidx[b] * 16;
  for (int j = 0; j < 8; ++j)  d[j]     = (__bf16)xh[((size_t)b * TSTEPS + t) * 8 + j];
  for (int j = 0; j < 16; ++j) d[8 + j] = (__bf16)e[j];
  for (int j = 24; j < 32; ++j) d[j]    = (__bf16)0.f;
}

__global__ void k_build_decx(const float* xf, const float* emb, const int* tidx,
                             const float* y0, __bf16* X) {
  int i = blockIdx.x * TPB + threadIdx.x;
  if (i >= TSTEPS * BATCH) return;
  int t = i / BATCH, b = i - t * BATCH;
  __bf16* d = X + (size_t)i * 32;
  const float* e = emb + (size_t)tidx[b] * 16;
  d[0] = (__bf16)((t == 0) ? y0[b] : 0.f);      // y slot, overwritten autoregressively
  for (int j = 0; j < 6; ++j)  d[1 + j] = (__bf16)xf[((size_t)b * TSTEPS + t) * 6 + j];
  for (int j = 0; j < 16; ++j) d[7 + j] = (__bf16)e[j];
  for (int j = 23; j < 32; ++j) d[j]    = (__bf16)0.f;
}

__global__ void k_init(__bf16* h0, __bf16* h1, unsigned* bar) {
  int i = blockIdx.x * TPB + threadIdx.x;
  if (i == 0) *bar = 0u;
  for (; i < BATCH * HID; i += gridDim.x * TPB) {
    h0[i] = (__bf16)0.f; h1[i] = (__bf16)0.f;
  }
}

// --------------------------------- launch ----------------------------------

extern "C" void kernel_launch(void* const* d_in, const int* in_sizes, int n_in,
                              void* d_out, int out_size, void* d_ws, size_t ws_size,
                              hipStream_t stream) {
  (void)in_sizes; (void)n_in; (void)out_size; (void)ws_size;
  const float* x_hist   = (const float*)d_in[0];
  const float* x_future = (const float*)d_in[1];
  const float* y0       = (const float*)d_in[2];
  const int*   tidx     = (const int*)d_in[3];
  const float* emb      = (const float*)d_in[5];
  const float* eWih0    = (const float*)d_in[6];
  const float* eWhh0    = (const float*)d_in[7];
  const float* eBih0    = (const float*)d_in[8];
  const float* eBhh0    = (const float*)d_in[9];
  const float* eWih1    = (const float*)d_in[10];
  const float* eWhh1    = (const float*)d_in[11];
  const float* eBih1    = (const float*)d_in[12];
  const float* eBhh1    = (const float*)d_in[13];
  const float* dWih0    = (const float*)d_in[14];
  const float* dWhh0    = (const float*)d_in[15];
  const float* dBih0    = (const float*)d_in[16];
  const float* dBhh0    = (const float*)d_in[17];
  const float* dWih1    = (const float*)d_in[18];
  const float* dWhh1    = (const float*)d_in[19];
  const float* dBih1    = (const float*)d_in[20];
  const float* dBhh1    = (const float*)d_in[21];
  const float* projW    = (const float*)d_in[22];
  const float* projB    = (const float*)d_in[23];

  char* w = (char*)d_ws;
  auto alloc = [&](size_t bytes) {
    void* p = (void*)w; w += (bytes + 255) & ~(size_t)255; return p;
  };
  __bf16* encWih0p = (__bf16*)alloc((size_t)G4H * 32 * 2);
  __bf16* encWhh0  = (__bf16*)alloc((size_t)G4H * HID * 2);
  __bf16* encWih1  = (__bf16*)alloc((size_t)G4H * HID * 2);
  __bf16* encWhh1  = (__bf16*)alloc((size_t)G4H * HID * 2);
  __bf16* decWih0p = (__bf16*)alloc((size_t)G4H * 32 * 2);
  __bf16* decWhh0  = (__bf16*)alloc((size_t)G4H * HID * 2);
  __bf16* decWih1  = (__bf16*)alloc((size_t)G4H * HID * 2);
  __bf16* decWhh1  = (__bf16*)alloc((size_t)G4H * HID * 2);
  __bf16* encX     = (__bf16*)alloc((size_t)TSTEPS * BATCH * 32 * 2);
  __bf16* decX     = (__bf16*)alloc((size_t)TSTEPS * BATCH * 32 * 2);
  __bf16* h0bf     = (__bf16*)alloc((size_t)BATCH * HID * 2);
  __bf16* h1bf     = (__bf16*)alloc((size_t)BATCH * HID * 2);
  float*  G        = (float*)alloc((size_t)NBLK * BATCH * 16 * 4);
  float*  ypart    = (float*)alloc((size_t)NBLK * BATCH * 4);
  float*  encB0    = (float*)alloc(G4H * 4);
  float*  encB1    = (float*)alloc(G4H * 4);
  float*  decB0    = (float*)alloc(G4H * 4);
  float*  decB1    = (float*)alloc(G4H * 4);
  unsigned* bar    = (unsigned*)alloc(256);

  // weight conversion (f32 -> bf16, K padded to 32 for layer-0 Wih)
  k_cvt_pad<<<1024, TPB, 0, stream>>>(eWih0, encWih0p, G4H, 24, 32);
  k_cvt_pad<<<1024, TPB, 0, stream>>>(eWhh0, encWhh0, G4H, HID, HID);
  k_cvt_pad<<<1024, TPB, 0, stream>>>(eWih1, encWih1, G4H, HID, HID);
  k_cvt_pad<<<1024, TPB, 0, stream>>>(eWhh1, encWhh1, G4H, HID, HID);
  k_cvt_pad<<<1024, TPB, 0, stream>>>(dWih0, decWih0p, G4H, 23, 32);
  k_cvt_pad<<<1024, TPB, 0, stream>>>(dWhh0, decWhh0, G4H, HID, HID);
  k_cvt_pad<<<1024, TPB, 0, stream>>>(dWih1, decWih1, G4H, HID, HID);
  k_cvt_pad<<<1024, TPB, 0, stream>>>(dWhh1, decWhh1, G4H, HID, HID);
  // fused biases
  k_add<<<G4H / TPB, TPB, 0, stream>>>(eBih0, eBhh0, encB0, G4H);
  k_add<<<G4H / TPB, TPB, 0, stream>>>(eBih1, eBhh1, encB1, G4H);
  k_add<<<G4H / TPB, TPB, 0, stream>>>(dBih0, dBhh0, decB0, G4H);
  k_add<<<G4H / TPB, TPB, 0, stream>>>(dBih1, dBhh1, decB1, G4H);
  // padded bf16 input sequences
  k_build_encx<<<(TSTEPS * BATCH) / TPB, TPB, 0, stream>>>(x_hist, emb, tidx, encX);
  k_build_decx<<<(TSTEPS * BATCH) / TPB, TPB, 0, stream>>>(x_future, emb, tidx, y0, decX);
  // zero state + reset grid barrier (every launch: ws not re-poisoned)
  k_init<<<NBLK, TPB, 0, stream>>>(h0bf, h1bf, bar);

  Params p;
  p.encWih0p = encWih0p; p.encWhh0 = encWhh0; p.encWih1 = encWih1; p.encWhh1 = encWhh1;
  p.decWih0p = decWih0p; p.decWhh0 = decWhh0; p.decWih1 = decWih1; p.decWhh1 = decWhh1;
  p.encX = encX; p.decX = decX; p.h0bf = h0bf; p.h1bf = h1bf;
  p.G = G; p.ypart = ypart;
  p.encB0 = encB0; p.encB1 = encB1; p.decB0 = decB0; p.decB1 = decB1;
  p.projW = projW; p.projB = projB;
  p.out = (float*)d_out; p.bar = bar;

  k_lstm<<<NBLK, TPB, 0, stream>>>(p);
}
